// BipartiteGNNConvFactorToVariable_3728031613009
// MI455X (gfx1250) — compile-verified, matrix operand order checked by reading, not run on hardware
//
#include <hip/hip_runtime.h>
#include <hip/hip_bf16.h>

#define DDIM 128
#define KDIM 256
#define TILE_ROWS 128
#define LDS_STRIDE 264   // 256 + 8 halves pad: keeps 16B alignment, breaks bank periodicity

typedef __attribute__((ext_vector_type(16))) __bf16 v16bf;
typedef __attribute__((ext_vector_type(8)))  __bf16 v8bf;
typedef __attribute__((ext_vector_type(4)))  __bf16 v4bf;
typedef __attribute__((ext_vector_type(8)))  float  v8f;

// Native conversion: backend emits packed v_cvt (round-to-nearest-even).
static __device__ __forceinline__ __bf16 f2bf(float f) { return (__bf16)f; }

// Load one 16x32-bf16 operand fragment for this lane from an LDS row laid out
// [row][K] contiguous-in-K. Per CDNA5 ISA 16-bit operand layout: lanes 0-15
// hold K = k0+0..7 and k0+16..23; lanes 16-31 hold K = k0+8..15 and k0+24..31.
// Both halves are contiguous 16B runs -> two ds_load_b128.
static __device__ __forceinline__ v16bf ldfrag(const __bf16* rowp, int k0, int koff) {
  v8bf lo = *(const v8bf*)(rowp + k0 + koff);
  v8bf hi = *(const v8bf*)(rowp + k0 + 16 + koff);
  return __builtin_shufflevector(lo, hi, 0,1,2,3,4,5,6,7,8,9,10,11,12,13,14,15);
}

// Convert a float4 to 4 packed bf16 and store as one 8-byte LDS write.
static __device__ __forceinline__ void store_pk4(__bf16* dst, float4 v) {
  v4bf p;
  p[0] = f2bf(v.x); p[1] = f2bf(v.y); p[2] = f2bf(v.z); p[3] = f2bf(v.w);
  *(v4bf*)dst = p;                      // ds_store_b64
}

__global__ void __launch_bounds__(256)
zero_f32_kernel(float* __restrict__ p, unsigned long long n) {
  unsigned long long i = (unsigned long long)blockIdx.x * blockDim.x + threadIdx.x;
  if (i < n) p[i] = 0.0f;
}

// Fused: gather x_i/x_j -> bf16 LDS tile -> WMMA message GEMM -> bias/relu ->
// scatter-add (f32 atomics) into aggr[v_to_f]. Never materializes m in HBM.
__global__ void __launch_bounds__(256)
edge_msg_scatter_kernel(const float* __restrict__ vars,
                        const float* __restrict__ facts,
                        const int*   __restrict__ vtof,
                        const int*   __restrict__ ftov,
                        const float* __restrict__ Wmsg,   // [257][128] row-major, row 256 dead (ea==0)
                        const float* __restrict__ bmsg,   // [128]
                        float*       __restrict__ aggr,   // [NV][128]
                        int E)
{
  __shared__ __bf16 Wt[DDIM][LDS_STRIDE];       // Wt[n][k]  (B operand, N-major)
  __shared__ __bf16 A[TILE_ROWS][LDS_STRIDE];   // A[edge][k] = [x_i | x_j]
  __shared__ int    vidx[TILE_ROWS];

  const int t    = threadIdx.x;
  const int lane = t & 31;
  const int wv   = t >> 5;                      // 8 waves per workgroup

  // One-time per-workgroup: load + transpose W_msg[0:256] to bf16 in LDS.
  for (int i = t; i < KDIM * DDIM; i += 256) {
    int k = i >> 7, n = i & (DDIM - 1);
    Wt[n][k] = f2bf(Wmsg[i]);
  }

  const int numTiles = (E + TILE_ROWS - 1) / TILE_ROWS;
  const __bf16 zh = f2bf(0.0f);

  for (int tile = blockIdx.x; tile < numTiles; tile += gridDim.x) {
    const int base = tile * TILE_ROWS;

    // Prefetch next tile's edge indices into cache while this tile computes.
    {
      const int nbase = base + (int)gridDim.x * TILE_ROWS;
      if (nbase + t < E) {
        __builtin_prefetch(vtof + nbase + t, 0, 1);
        __builtin_prefetch(ftov + nbase + t, 0, 1);
      }
    }

    __syncthreads();                            // also covers the W-load on iter 0

    if (t < TILE_ROWS) {
      int e = base + t;
      vidx[t] = (e < E) ? vtof[e] : -1;
    }
    __syncthreads();

    // Stage A tile: 2 threads per edge row (half 0 = x_i, half 1 = x_j).
    {
      const int r    = t >> 1;
      const int half = t & 1;
      const int e    = base + r;
      __bf16* dst = &A[r][half * DDIM];
      if (e < E) {
        const float* src = half ? (facts + (size_t)ftov[e] * DDIM)
                                : (vars  + (size_t)vtof[e] * DDIM);
        #pragma unroll
        for (int c = 0; c < DDIM; c += 4) {
          store_pk4(dst + c, *(const float4*)(src + c));
        }
      } else {
        #pragma unroll
        for (int c = 0; c < DDIM; ++c) dst[c] = zh;
      }
    }
    __syncthreads();

    // Each wave: 16 edges x 128 outputs. 8 K-steps x 8 N-tiles = 64 WMMA.
    const int rowA = (wv << 4) + (lane & 15);
    const int koff = (lane >> 4) << 3;
    v8f acc[8] = {};
    #pragma unroll
    for (int k0 = 0; k0 < KDIM; k0 += 32) {
      v16bf a = ldfrag(&A[rowA][0], k0, koff);
      #pragma unroll
      for (int nt = 0; nt < 8; ++nt) {
        v16bf b = ldfrag(&Wt[nt * 16 + (lane & 15)][0], k0, koff);
        acc[nt] = __builtin_amdgcn_wmma_f32_16x16x32_bf16(
            false, a, false, b, (short)0, acc[nt], false, false);
      }
    }

    // Epilogue: bias + relu + scatter-add to aggr (non-returning f32 atomics).
    #pragma unroll
    for (int nt = 0; nt < 8; ++nt) {
      const int col = nt * 16 + (lane & 15);
      const float bias = bmsg[col];
      #pragma unroll
      for (int r = 0; r < 8; ++r) {
        const int m  = r + ((lane >> 4) << 3);  // C/D layout: lanes 16-31 hold M+8
        const int vi = vidx[(wv << 4) + m];
        if (vi >= 0) {
          float val = acc[nt][r] + bias;
          val = val > 0.0f ? val : 0.0f;
          (void)__hip_atomic_fetch_add(&aggr[(size_t)vi * DDIM + col], val,
                                       __ATOMIC_RELAXED, __HIP_MEMORY_SCOPE_AGENT);
        }
      }
    }
  }
}

// Combine: h = relu([vars | aggr] @ W_comb + b_comb); out = vars + h
__global__ void __launch_bounds__(256)
combine_kernel(const float* __restrict__ vars,
               const float* __restrict__ aggr,
               const float* __restrict__ Wcomb,  // [256][128] row-major
               const float* __restrict__ bcomb,  // [128]
               float*       __restrict__ out,
               int NV)
{
  __shared__ __bf16 Wt[DDIM][LDS_STRIDE];
  __shared__ __bf16 A[TILE_ROWS][LDS_STRIDE];

  const int t    = threadIdx.x;
  const int lane = t & 31;
  const int wv   = t >> 5;

  for (int i = t; i < KDIM * DDIM; i += 256) {
    int k = i >> 7, n = i & (DDIM - 1);
    Wt[n][k] = f2bf(Wcomb[i]);
  }

  const int numTiles = (NV + TILE_ROWS - 1) / TILE_ROWS;
  const __bf16 zh = f2bf(0.0f);

  for (int tile = blockIdx.x; tile < numTiles; tile += gridDim.x) {
    const int base = tile * TILE_ROWS;
    __syncthreads();

    {
      const int r    = t >> 1;
      const int half = t & 1;
      const int node = base + r;
      __bf16* dst = &A[r][half * DDIM];
      if (node < NV) {
        const float* src = half ? (aggr + (size_t)node * DDIM)
                                : (vars + (size_t)node * DDIM);
        #pragma unroll
        for (int c = 0; c < DDIM; c += 4) {
          store_pk4(dst + c, *(const float4*)(src + c));
        }
      } else {
        #pragma unroll
        for (int c = 0; c < DDIM; ++c) dst[c] = zh;
      }
    }
    __syncthreads();

    const int rowA = (wv << 4) + (lane & 15);
    const int koff = (lane >> 4) << 3;
    v8f acc[8] = {};
    #pragma unroll
    for (int k0 = 0; k0 < KDIM; k0 += 32) {
      v16bf a = ldfrag(&A[rowA][0], k0, koff);
      #pragma unroll
      for (int nt = 0; nt < 8; ++nt) {
        v16bf b = ldfrag(&Wt[nt * 16 + (lane & 15)][0], k0, koff);
        acc[nt] = __builtin_amdgcn_wmma_f32_16x16x32_bf16(
            false, a, false, b, (short)0, acc[nt], false, false);
      }
    }

    #pragma unroll
    for (int nt = 0; nt < 8; ++nt) {
      const int col  = nt * 16 + (lane & 15);
      const float bias = bcomb[col];
      #pragma unroll
      for (int r = 0; r < 8; ++r) {
        const int m    = r + ((lane >> 4) << 3);
        const int node = base + (wv << 4) + m;
        if (node < NV) {
          float val = acc[nt][r] + bias;
          val = val > 0.0f ? val : 0.0f;
          out[(size_t)node * DDIM + col] = vars[(size_t)node * DDIM + col] + val;
        }
      }
    }
  }
}

extern "C" void kernel_launch(void* const* d_in, const int* in_sizes, int n_in,
                              void* d_out, int out_size, void* d_ws, size_t ws_size,
                              hipStream_t stream) {
  const float* vars  = (const float*)d_in[0];
  const float* facts = (const float*)d_in[1];
  const int*   vtof  = (const int*)d_in[2];
  const int*   ftov  = (const int*)d_in[3];
  // d_in[4] = edge_attr: forward uses zeros_like -> unused
  const float* Wmsg  = (const float*)d_in[5];
  const float* bmsg  = (const float*)d_in[6];
  const float* Wcomb = (const float*)d_in[7];
  const float* bcomb = (const float*)d_in[8];

  const int E  = in_sizes[2];
  const int NV = in_sizes[0] / DDIM;

  float* aggr = (float*)d_ws;                    // [NV][128] f32 scratch (51.2 MB)
  float* out  = (float*)d_out;

  const unsigned long long aggrN = (unsigned long long)NV * DDIM;
  zero_f32_kernel<<<(unsigned)((aggrN + 255) / 256), 256, 0, stream>>>(aggr, aggrN);
  edge_msg_scatter_kernel<<<512, 256, 0, stream>>>(vars, facts, vtof, ftov,
                                                   Wmsg, bmsg, aggr, E);
  combine_kernel<<<512, 256, 0, stream>>>(vars, aggr, Wcomb, bcomb, out, NV);
}
